// PoincareEmbeddings_72301479461136
// MI455X (gfx1250) — matrix operand order repaired and checked by reading, not compiled, not run on hardware
//
#include <hip/hip_runtime.h>
#include <hip/hip_bf16.h>

typedef _Float16 v16h __attribute__((ext_vector_type(16)));
typedef _Float16 v8h  __attribute__((ext_vector_type(8)));
typedef float    v8f  __attribute__((ext_vector_type(8)));

constexpr int V_   = 100000;
constexpr int P_   = 300;     // custom (Poincare) dim
constexpr int E_   = 256;     // output dim
constexpr int BL   = 1024 * 256;  // total tokens
constexpr int PK   = 320;     // P padded to multiple of 32 (WMMA K)
constexpr int TM   = 64;      // tokens per workgroup (4 M-tiles)
constexpr int LDA  = 328;     // LDS A row stride in halves (bank spread, 16B aligned)

// ---------------------------------------------------------------------------
// Kernel 0: convert lin_w [E][P] f32 -> [E][PK] f16 in workspace (zero-pad K)
// ---------------------------------------------------------------------------
__global__ void poincare_convert_linw(const float* __restrict__ lin_w,
                                      _Float16* __restrict__ wsh) {
  int idx = blockIdx.x * blockDim.x + threadIdx.x;   // 0 .. E_*PK-1
  int e = idx / PK;
  int k = idx - e * PK;
  float v = (k < P_) ? lin_w[e * P_ + k] : 0.0f;
  wsh[idx] = (_Float16)v;
}

// ---------------------------------------------------------------------------
// Fused kernel: hyperbolic embedding (VALU) -> LDS f16 panel -> WMMA GEMM
//               -> + lin_b + regular_w gather -> out
// One workgroup: 256 threads (8 waves), 64 tokens, full 256 output columns.
// ---------------------------------------------------------------------------
__global__ __launch_bounds__(256) void poincare_fused(
    const int*   __restrict__ x,
    const int*   __restrict__ vocab_to_custom,
    const int*   __restrict__ vocab_to_regular,
    const float* __restrict__ custom_fixed_w,   // [(K+1), P]
    const float* __restrict__ custom_train_w,   // [(K+1), P]
    const float* __restrict__ regular_w,        // [V-K, E]
    const _Float16* __restrict__ lin_w_h,       // [E][PK] f16 (workspace)
    const float* __restrict__ lin_b,            // [E]
    float*       __restrict__ out)              // [BL, E]
{
  __shared__ __attribute__((aligned(64))) _Float16 sA[TM * LDA];
  __shared__ int s_rm[TM];

  const int tid    = threadIdx.x;
  const int lane   = tid & 31;
  const int w      = tid >> 5;            // wave id 0..7
  const int wgbase = blockIdx.x * TM;

  // ---------------- Phase 1: per-token logmap0 + mobius_add -> LDS --------
  for (int i = 0; i < 8; ++i) {
    const int trow = w * 8 + i;
    const int tok  = wgbase + trow;
    const int xv   = x[tok];
    const int cm   = vocab_to_custom[xv];   // 0 for non-custom (zero rows)
    if (lane == 0) s_rm[trow] = vocab_to_regular[xv];

    const float* fptr = custom_fixed_w + (size_t)cm * P_;
    const float* tptr = custom_train_w + (size_t)cm * P_;

    float f[10], t[10];
    float nf = 0.f;
    #pragma unroll
    for (int j = 0; j < 10; ++j) {
      int p = lane + 32 * j;
      f[j] = (p < P_) ? fptr[p] : 0.f;
      nf += f[j] * f[j];
    }
    #pragma unroll
    for (int m = 16; m >= 1; m >>= 1) nf += __shfl_xor(nf, m, 32);

    float norm  = sqrtf(nf);
    float scale = (norm > 0.f) ? (atanhf(norm) / norm) : 1.0f;  // logmap0 scale
    float x2    = scale * scale * nf;                            // |ft|^2

    float y2 = 0.f, xy = 0.f;
    #pragma unroll
    for (int j = 0; j < 10; ++j) {
      int p = lane + 32 * j;
      t[j] = (p < P_) ? tptr[p] : 0.f;
      float ft = scale * f[j];
      y2 += t[j] * t[j];
      xy += ft   * t[j];
    }
    #pragma unroll
    for (int m = 16; m >= 1; m >>= 1) {
      y2 += __shfl_xor(y2, m, 32);
      xy += __shfl_xor(xy, m, 32);
    }

    float ca  = 1.f + 2.f * xy + y2;        // coeff on ft
    float cb  = 1.f - x2;                   // coeff on t
    float inv = 1.f / fmaxf(1.f + 2.f * xy + x2 * y2, 1e-15f);

    #pragma unroll
    for (int j = 0; j < 10; ++j) {
      int p = lane + 32 * j;               // covers 0..319
      float ce = (p < P_) ? (ca * (scale * f[j]) + cb * t[j]) * inv : 0.f;
      sA[trow * LDA + p] = (_Float16)ce;
    }
  }
  __syncthreads();

  // ---------------- Phase 2: WMMA GEMM  D[64x256] = A[64x320] x B[320x256] --
  v8f acc[4][2];
  #pragma unroll
  for (int m = 0; m < 4; ++m) { acc[m][0] = {}; acc[m][1] = {}; }

  const int nt0   = 2 * w;                 // this wave's first N-tile
  const int eCol  = (lane & 15);           // column-in-tile for B / C / D
  const int bhi   = (lane >> 4);           // 0: K lo half, 1: K hi half
  const int ahalf = bhi ? 16 : 0;          // A frag K sub-offset (contiguous K, same as B)

  for (int kk = 0; kk < PK; kk += 32) {
    // B fragments: lane holds 16 contiguous K halves of its output column
    const _Float16* b0p = lin_w_h + (size_t)(nt0 * 16 + eCol) * PK + kk + 16 * bhi;
    const _Float16* b1p = b0p + (size_t)16 * PK;
    if (kk + 32 < PK) __builtin_prefetch(b0p + 32, 0, 0);   // global_prefetch
    v16h bf0 = *(const v16h*)b0p;
    v16h bf1 = *(const v16h*)b1p;

    #pragma unroll
    for (int m = 0; m < 4; ++m) {
      // A fragment, contiguous K order matching B:
      //   lanes 0-15 : row=lane,    K = kk+{0..15}
      //   lanes 16-31: row=lane-16, K = kk+{16..31}
      const _Float16* ap = sA + (size_t)(m * 16 + eCol) * LDA + kk + ahalf;
      v8h a_lo = *(const v8h*)ap;           // ds_read_b128
      v8h a_hi = *(const v8h*)(ap + 8);     // ds_read_b128
      v16h af;
      #pragma unroll
      for (int q = 0; q < 8; ++q) { af[q] = a_lo[q]; af[8 + q] = a_hi[q]; }

      acc[m][0] = __builtin_amdgcn_wmma_f32_16x16x32_f16(
          false, af, false, bf0, (short)0, acc[m][0], false, false);
      acc[m][1] = __builtin_amdgcn_wmma_f32_16x16x32_f16(
          false, af, false, bf1, (short)0, acc[m][1], false, false);
    }
  }

  // ---------------- Epilogue: + lin_b + regular embedding gather ----------
  // C/D 16x16 f32 layout: lane = column; VGPR r -> row r (lanes 0-15) / 8+r
  const int mrow = bhi ? 8 : 0;
  #pragma unroll
  for (int j = 0; j < 2; ++j) {
    const int e    = (nt0 + j) * 16 + eCol;
    const float bv = lin_b[e];
    #pragma unroll
    for (int m = 0; m < 4; ++m) {
      #pragma unroll
      for (int r = 0; r < 8; ++r) {
        const int trow = m * 16 + mrow + r;
        const int tok  = wgbase + trow;
        const float rv = regular_w[(size_t)s_rm[trow] * E_ + e];
        out[(size_t)tok * E_ + e] = acc[m][j][r] + bv + rv;
      }
    }
  }
}

// ---------------------------------------------------------------------------
extern "C" void kernel_launch(void* const* d_in, const int* in_sizes, int n_in,
                              void* d_out, int out_size, void* d_ws, size_t ws_size,
                              hipStream_t stream) {
  (void)in_sizes; (void)n_in; (void)out_size; (void)ws_size;
  const int*   x    = (const int*)  d_in[0];
  // d_in[1] = custom_indices (unused: vocab_to_custom already encodes the mask)
  const int*   v2c  = (const int*)  d_in[2];
  const int*   v2r  = (const int*)  d_in[3];
  const float* cfw  = (const float*)d_in[4];
  const float* ctw  = (const float*)d_in[5];
  const float* regw = (const float*)d_in[6];
  const float* linw = (const float*)d_in[7];
  const float* linb = (const float*)d_in[8];
  float*    out = (float*)d_out;
  _Float16* wsh = (_Float16*)d_ws;   // needs E_*PK*2 = 160 KB

  poincare_convert_linw<<<(E_ * PK) / 256, 256, 0, stream>>>(linw, wsh);
  poincare_fused<<<BL / TM, 256, 0, stream>>>(x, v2c, v2r, cfw, ctw, regw,
                                              wsh, linb, out);
}
